// Encoder_60344290508892
// MI455X (gfx1250) — matrix operand order repaired and checked by reading, not compiled, not run on hardware
//
#include <hip/hip_runtime.h>
#include <hip/hip_bf16.h>
#include <cstdint>

#define L_SEQ 256
#define BATCH 64
#define DIM   1024
#define LBD   (L_SEQ * BATCH * DIM)   // 16,777,216 elements
#define BD    (BATCH * DIM)           // 65,536 elements

typedef __attribute__((ext_vector_type(16))) __bf16 v16bf;
typedef __attribute__((ext_vector_type(8)))  float  v8f;

union Frag { v16bf v; uint4 q[2]; };

__device__ __forceinline__ unsigned short f2bf(float f) {
  unsigned int u = __float_as_uint(f);
  u += 0x7fffu + ((u >> 16) & 1u);          // round-to-nearest-even
  return (unsigned short)(u >> 16);
}
__device__ __forceinline__ float sigmoid_f(float x) {
  return 1.0f / (1.0f + __expf(-x));
}
__device__ __forceinline__ float tanh_f(float x) {
  float e = __expf(-2.0f * x);
  return (1.0f - e) / (1.0f + e);
}

// issue one async global->LDS 16-byte transfer (gfx1250, tracked by ASYNCcnt)
__device__ __forceinline__ void async_b128(unsigned lds_addr, const void* gptr) {
  asm volatile("global_load_async_to_lds_b128 %0, %1, off"
               :: "v"(lds_addr), "v"((unsigned long long)(uintptr_t)gptr)
               : "memory");
}

// ---------------- embedding gather: x = emb[xs], fp32 + bf16 copies ----------
__global__ void embed_kernel(const int* __restrict__ xs, const float* __restrict__ emb,
                             float* __restrict__ x32, unsigned short* __restrict__ x16) {
  int row = blockIdx.x;                        // 0 .. L*B-1
  int tok = xs[row];
  const float* src = emb + (size_t)tok * DIM;
  float* d32 = x32 + (size_t)row * DIM;
  unsigned short* d16 = x16 + (size_t)row * DIM;
  for (int i = threadIdx.x; i < DIM; i += blockDim.x) {
    float v = src[i];
    d32[i] = v;
    d16[i] = f2bf(v);
  }
}

// ---------------- weight transpose + fp32->bf16: Wt[n,k] = bf16(W[k,n]) ------
__global__ void wconv_kernel(const float* __restrict__ W, unsigned short* __restrict__ Wt) {
  int idx = blockIdx.x * blockDim.x + threadIdx.x;   // over D*D
  int k = idx >> 10;
  int n = idx & (DIM - 1);
  Wt[(size_t)n * DIM + k] = f2bf(W[idx]);
}

__global__ void zero_c_kernel(float* __restrict__ c32, unsigned short* __restrict__ c16) {
  int idx = blockIdx.x * blockDim.x + threadIdx.x;
  if (idx < BD) { c32[idx] = 0.0f; c16[idx] = 0; }
}

// ---------------- projection GEMM (LDS-tiled, async double-buffered) ---------
// C[M,N] = A(bf16)[M,K] x Bt(bf16)[N,K]^T + bias ; M=16384, N=K=1024
// block tile 128(M) x 64(N), K staged in chunks of 64, 8 waves as 4(M) x 2(N),
// each wave computes a 32x32 macro-tile = 2x2 WMMA tiles.
#define BM 128
#define BN 64
#define KC 64
#define NSTAGE (DIM / KC)   // 16

__global__ __launch_bounds__(256) void proj_gemm_kernel(
    const unsigned short* __restrict__ A,    // [M,K] bf16 row-major
    const unsigned short* __restrict__ Bt,   // [N,K] bf16 (weight pre-transposed)
    const float* __restrict__ bias,          // [N]
    float* __restrict__ C)                   // [M,N] fp32
{
  __shared__ unsigned short Asmem[2][BM * KC];   // 2 x 16 KB
  __shared__ unsigned short Bsmem[2][BN * KC];   // 2 x  8 KB

  const int tid  = threadIdx.x;
  const int wave = tid >> 5;
  const int lane = tid & 31;
  const int hi   = lane >> 4;
  const int l15  = lane & 15;
  const int wm   = wave >> 1;                 // 0..3
  const int wn   = wave & 1;                  // 0..1
  const int bn   = blockIdx.x & (DIM / BN - 1);   // 0..15
  const int bm   = blockIdx.x >> 4;               // 0..127

  const unsigned short* Ag = A  + (size_t)(bm * BM) * DIM;   // + stage*KC cols
  const unsigned short* Bg = Bt + (size_t)(bn * BN) * DIM;

  // per-stage async fill: A slab 128x64 (8 b128-segs/row), B slab 64x64
  auto fill = [&](int stage, int buf) {
    const unsigned short* Ags = Ag + stage * KC;
    const unsigned short* Bgs = Bg + stage * KC;
    #pragma unroll
    for (int i = 0; i < 4; ++i) {                  // 1024 segs / 256 thr
      int idx = i * 256 + tid;
      int row = idx >> 3, seg = idx & 7;
      async_b128((unsigned)(uintptr_t)&Asmem[buf][row * KC + seg * 8],
                 Ags + (size_t)row * DIM + seg * 8);
    }
    #pragma unroll
    for (int i = 0; i < 2; ++i) {                  // 512 segs / 256 thr
      int idx = i * 256 + tid;
      int row = idx >> 3, seg = idx & 7;
      async_b128((unsigned)(uintptr_t)&Bsmem[buf][row * KC + seg * 8],
                 Bgs + (size_t)row * DIM + seg * 8);
    }
  };

  v8f acc[2][2] = {};
  fill(0, 0);

  for (int s = 0; s < NSTAGE; ++s) {
    if (s + 1 < NSTAGE) {
      fill(s + 1, (s + 1) & 1);
      // 6 transfers of stage s+1 may remain in flight; stage s's are retired
      asm volatile("s_wait_asynccnt 6" ::: "memory");
    } else {
      asm volatile("s_wait_asynccnt 0" ::: "memory");
    }
    __syncthreads();

    const unsigned short* As = &Asmem[s & 1][0];
    const unsigned short* Bs = &Bsmem[s & 1][0];
    #pragma unroll
    for (int kts = 0; kts < KC / 32; ++kts) {
      int ko = kts * 32;
      Frag a0, a1, b0, b1;
      // A 16x32 frag: lanes0-15 row=l15 K{0..7,16..23}; lanes16-31 K{8..15,24..31}
      {
        const unsigned short* p0 = As + (wm * 32 + l15) * KC + ko + hi * 8;
        const unsigned short* p1 = As + (wm * 32 + 16 + l15) * KC + ko + hi * 8;
        a0.q[0] = *(const uint4*)(p0);
        a0.q[1] = *(const uint4*)(p0 + 16);
        a1.q[0] = *(const uint4*)(p1);
        a1.q[1] = *(const uint4*)(p1 + 16);
      }
      // B 32x16 frag: lane col=l15; lanes0-15 K{0..7,16..23}; lanes16-31 K{8..15,24..31}
      {
        const unsigned short* p0 = Bs + (wn * 32 + l15) * KC + ko + hi * 8;
        const unsigned short* p1 = Bs + (wn * 32 + 16 + l15) * KC + ko + hi * 8;
        b0.q[0] = *(const uint4*)(p0);
        b0.q[1] = *(const uint4*)(p0 + 16);
        b1.q[0] = *(const uint4*)(p1);
        b1.q[1] = *(const uint4*)(p1 + 16);
      }
      acc[0][0] = __builtin_amdgcn_wmma_f32_16x16x32_bf16(false, a0.v, false, b0.v,
                                                          (short)0, acc[0][0], false, false);
      acc[0][1] = __builtin_amdgcn_wmma_f32_16x16x32_bf16(false, a0.v, false, b1.v,
                                                          (short)0, acc[0][1], false, false);
      acc[1][0] = __builtin_amdgcn_wmma_f32_16x16x32_bf16(false, a1.v, false, b0.v,
                                                          (short)0, acc[1][0], false, false);
      acc[1][1] = __builtin_amdgcn_wmma_f32_16x16x32_bf16(false, a1.v, false, b1.v,
                                                          (short)0, acc[1][1], false, false);
    }
    __syncthreads();
  }

  // epilogue: D layout -> lane holds col l15, rows r + 8*hi
  #pragma unroll
  for (int ms = 0; ms < 2; ++ms) {
    #pragma unroll
    for (int ns = 0; ns < 2; ++ns) {
      int n  = bn * BN + wn * 32 + ns * 16 + l15;
      int m0 = bm * BM + wm * 32 + ms * 16 + hi * 8;
      float bv = bias[n];
      float* crow = C + (size_t)m0 * DIM + n;
      #pragma unroll
      for (int r = 0; r < 8; ++r)
        crow[(size_t)r * DIM] = acc[ms][ns][r] + bv;
    }
  }
}

// ---------------- one RAN recurrence step, fused GEMM pair + elementwise -----
// i = sigma(xi + c@Wic); f = sigma(xf + c@Wfc); c' = i*xc + f*c; h = tanh(c') + x
__global__ __launch_bounds__(256) void ran_step_kernel(
    const float* __restrict__ c32p, const unsigned short* __restrict__ c16p,
    float* __restrict__ c32n, unsigned short* __restrict__ c16n,
    const unsigned short* __restrict__ WicT, const unsigned short* __restrict__ WfcT,
    const float* __restrict__ xi_t, const float* __restrict__ xf_t,
    const float* __restrict__ xc_t, const float* __restrict__ xres_t,
    float* __restrict__ h32_t, unsigned short* __restrict__ h16_t)
{
  const int K = DIM, N = DIM;
  int wave = threadIdx.x >> 5;
  int lane = threadIdx.x & 31;
  int tile = blockIdx.x * 8 + wave;          // 256 tiles: 4 (M) x 64 (N)
  int mt = tile >> 6;                        // 0..3
  int nt = tile & 63;                        // 0..63
  int hi  = lane >> 4;
  int l15 = lane & 15;

  const unsigned short* arow = c16p + (size_t)(mt * 16 + l15) * K + hi * 8;
  const unsigned short* bir  = WicT + (size_t)(nt * 16 + l15) * K + hi * 8;
  const unsigned short* bfr  = WfcT + (size_t)(nt * 16 + l15) * K + hi * 8;

  v8f acci = {}, accf = {};
  #pragma unroll 2
  for (int kk = 0; kk < K; kk += 32) {
    Frag a, b1, b2;
    a.q[0]  = *(const uint4*)(arow + kk);
    a.q[1]  = *(const uint4*)(arow + kk + 16);
    b1.q[0] = *(const uint4*)(bir + kk);
    b1.q[1] = *(const uint4*)(bir + kk + 16);
    b2.q[0] = *(const uint4*)(bfr + kk);
    b2.q[1] = *(const uint4*)(bfr + kk + 16);
    acci = __builtin_amdgcn_wmma_f32_16x16x32_bf16(false, a.v, false, b1.v,
                                                   (short)0, acci, false, false);
    accf = __builtin_amdgcn_wmma_f32_16x16x32_bf16(false, a.v, false, b2.v,
                                                   (short)0, accf, false, false);
  }

  int n  = nt * 16 + l15;
  int m0 = mt * 16 + hi * 8;
  #pragma unroll
  for (int r = 0; r < 8; ++r) {
    size_t idx = (size_t)(m0 + r) * N + n;
    float iv = sigmoid_f(xi_t[idx] + acci[r]);
    float fv = sigmoid_f(xf_t[idx] + accf[r]);
    float cn = iv * xc_t[idx] + fv * c32p[idx];
    float h  = tanh_f(cn) + xres_t[idx];
    c32n[idx] = cn;
    c16n[idx] = f2bf(cn);
    h32_t[idx] = h;
    h16_t[idx] = f2bf(h);
  }
}

// -----------------------------------------------------------------------------
extern "C" void kernel_launch(void* const* d_in, const int* in_sizes, int n_in,
                              void* d_out, int out_size, void* d_ws, size_t ws_size,
                              hipStream_t stream) {
  (void)in_sizes; (void)n_in; (void)out_size; (void)ws_size;

  const int*   xs  = (const int*)d_in[0];
  const float* emb = (const float*)d_in[1];
  // weight order per direction: Wix, Wic, Wfx, Wfc, Wcx
  const float* Wf32[10];
  for (int j = 0; j < 5; ++j) Wf32[j]     = (const float*)d_in[2  + j];
  for (int j = 0; j < 5; ++j) Wf32[5 + j] = (const float*)d_in[10 + j];
  const float* bias_[6] = {
    (const float*)d_in[7],  (const float*)d_in[8],  (const float*)d_in[9],    // fw bi,bf,bc
    (const float*)d_in[15], (const float*)d_in[16], (const float*)d_in[17] }; // bw bi,bf,bc

  // ---- workspace carve-up ----
  char* ws = (char*)d_ws;
  size_t off = 0;
  auto alloc = [&](size_t bytes) -> void* {
    void* p = ws + off; off += (bytes + 255) & ~(size_t)255; return p;
  };
  float*          X32[2]; unsigned short* X16[2];
  X32[0] = (float*)alloc((size_t)LBD * 4);
  X32[1] = (float*)alloc((size_t)LBD * 4);
  X16[0] = (unsigned short*)alloc((size_t)LBD * 2);
  X16[1] = (unsigned short*)alloc((size_t)LBD * 2);
  float* XI = (float*)alloc((size_t)LBD * 4);
  float* XF = (float*)alloc((size_t)LBD * 4);
  float* XC = (float*)alloc((size_t)LBD * 4);
  float*          C32[2]; unsigned short* C16[2];
  C32[0] = (float*)alloc((size_t)BD * 4);
  C32[1] = (float*)alloc((size_t)BD * 4);
  C16[0] = (unsigned short*)alloc((size_t)BD * 2);
  C16[1] = (unsigned short*)alloc((size_t)BD * 2);
  unsigned short* Wt[10];
  for (int j = 0; j < 10; ++j) Wt[j] = (unsigned short*)alloc((size_t)DIM * DIM * 2);

  // ---- prologue: embed + weight convert + zero state ----
  embed_kernel<<<L_SEQ * BATCH, 256, 0, stream>>>(xs, emb, X32[0], X16[0]);
  for (int j = 0; j < 10; ++j)
    wconv_kernel<<<DIM * DIM / 256, 256, 0, stream>>>(Wf32[j], Wt[j]);
  zero_c_kernel<<<BD / 256, 256, 0, stream>>>(C32[0], C16[0]);

  // ---- 2 layers x (forward scan, backward scan); c threads through all scans
  const int proj_grid = (L_SEQ * BATCH / BM) * (DIM / BN);   // 128 * 16 = 2048
  int cur = 0;   // activation ping-pong
  int p   = 0;   // c parity
  for (int layer = 0; layer < 2; ++layer) {
    for (int dir = 0; dir < 2; ++dir) {
      int w0 = dir * 5;
      // hoisted projections xi/xf/xc for all time steps (16384 x 1024 each)
      proj_gemm_kernel<<<proj_grid, 256, 0, stream>>>(X16[cur], Wt[w0 + 0], bias_[dir*3 + 0], XI);
      proj_gemm_kernel<<<proj_grid, 256, 0, stream>>>(X16[cur], Wt[w0 + 2], bias_[dir*3 + 1], XF);
      proj_gemm_kernel<<<proj_grid, 256, 0, stream>>>(X16[cur], Wt[w0 + 4], bias_[dir*3 + 2], XC);
      int nxt = cur ^ 1;
      for (int s = 0; s < L_SEQ; ++s) {
        int t = dir ? (L_SEQ - 1 - s) : s;
        size_t o = (size_t)t * BD;
        ran_step_kernel<<<32, 256, 0, stream>>>(
            C32[p], C16[p], C32[p ^ 1], C16[p ^ 1],
            Wt[w0 + 1], Wt[w0 + 3],
            XI + o, XF + o, XC + o, X32[cur] + o,
            X32[nxt] + o, X16[nxt] + o);
        p ^= 1;
      }
      cur = nxt;
    }
  }
  hipMemcpyAsync(d_out, X32[cur], (size_t)LBD * 4, hipMemcpyDeviceToDevice, stream);
}
